// GraphBlock_49795850830076
// MI455X (gfx1250) — matrix operand order repaired and checked by reading, not compiled, not run on hardware
//
#include <hip/hip_runtime.h>
#include <hip/hip_bf16.h>
#include <cstdint>
#include <cstddef>

// ---------------------------------------------------------------------------
// MI455X (gfx1250) implementation. All dense algebra runs through
// v_wmma_f32_16x16x32_f16 (wave32, 16x16 tiles). Intermediates kept in f16
// (fits L2, halves HBM traffic vs f32); bias/ReLU/LeakyReLU/adjacency epilogues
// fused on the f32 accumulators. Tiles staged in LDS via async LDS-direct
// loads (ASYNCcnt) — confirmed lowering on this toolchain.
// ---------------------------------------------------------------------------

typedef _Float16 half_t;
typedef __attribute__((ext_vector_type(16))) _Float16 v16h;
typedef __attribute__((ext_vector_type(8)))  float    v8f;

#define AS1 __attribute__((address_space(1)))
#define AS3 __attribute__((address_space(3)))

// exact pointee type of the async-load builtin params (from hipcc diagnostic)
typedef int v4i_t __attribute__((__vector_size__(4 * sizeof(int))));
typedef AS1 v4i_t* gv4_t;
typedef AS3 v4i_t* lv4_t;

#if __has_builtin(__builtin_amdgcn_global_load_async_to_lds_b128)
#define ASYNC_LDS 1
#else
#define ASYNC_LDS 0
#endif

__device__ __forceinline__ void wait_async() {
#if __has_builtin(__builtin_amdgcn_s_wait_asynccnt)
  __builtin_amdgcn_s_wait_asynccnt(0);
#elif ASYNC_LDS
  asm volatile("s_wait_asynccnt 0" ::: "memory");
#endif
}

enum { EP_NONE = 0, EP_BIAS = 1, EP_BIAS_RELU = 2, EP_BIAS_LEAKY = 3, EP_ADJ = 4 };
enum { A_DENSE = 0, A_CONV = 1 };
enum { B_KN = 0, B_NK = 1 };

constexpr int GM = 64, GN = 64, GK = 32;
constexpr int LDP = 40;   // padded LDS row stride (f16): 80B rows, 16B-aligned frag reads

// ---------------------------------------------------------------------------
// WMMA GEMM core.  C[M,N] = A[M,K] * B + fused epilogue.
//  AMODE: dense f16 row-major, or branch-free implicit-im2col gather from a
//         halo-padded (Nseq, C, Lout+7) f16 feature map (k=8, pad=4 conv).
//  BMODE: B_KN = B is K x N row-major;  B_NK = B is N x K row-major (A*B^T).
//  CLOUT: compile-time conv Lout (strength-reduces all div/mod); 0 = dynamic.
//  Batched via blockIdx.z with element strides sA/sB/sC.
// ---------------------------------------------------------------------------
template<int AMODE, int BMODE, int CLOUT = 0>
__global__ __launch_bounds__(128) void k_gemm_wmma(
    const half_t* __restrict__ A, const half_t* __restrict__ B,
    const float* __restrict__ bias, void* __restrict__ Cv,
    int M, int N, int K, long sA, long sB, long sC,
    int ep, int store_half, int conv_store,
    int cLout, int cCin)
{
  __shared__ __align__(16) half_t As[GM * LDP];
  __shared__ __align__(16) half_t Bs[GN * LDP];

  const int z = blockIdx.z;
  const half_t* Ab = A + (long)z * sA;
  const half_t* Bb = B + (long)z * sB;
  const int m0 = blockIdx.x * GM;
  const int n0 = blockIdx.y * GN;
  const int tid  = threadIdx.x;
  const int lane = tid & 31, wave = tid >> 5;
  const int wm = (wave >> 1) * 32, wn = (wave & 1) * 32;
  const int fm = lane & 15, fh = lane >> 4;
  const int lout = CLOUT ? CLOUT : cLout;        // constant-folded for conv calls
  const bool fullM = (m0 + GM <= M);
  const bool fullN = (n0 + GN <= N);

  v8f acc[2][2] = {};

  const int nK = K / GK;
  for (int kk = 0; kk < nK; ++kk) {
    const int k0 = kk * GK;
    bool used_async = false;

    // ---- stage A tile into LDS: As[m][k] ----
    if (AMODE == A_DENSE) {
#if ASYNC_LDS
      if (fullM) {
        #pragma unroll
        for (int it = 0; it < 2; ++it) {
          int idx = (it * 128 + tid) * 8;
          int r = idx / GK, c = idx % GK;
          __builtin_amdgcn_global_load_async_to_lds_b128(
              (gv4_t)(AS1 void*)(void*)(Ab + (long)(m0 + r) * K + k0 + c),
              (lv4_t)(AS3 void*)(void*)&As[r * LDP + c], 0, 0);
        }
        used_async = true;
      } else
#endif
      {
        #pragma unroll
        for (int it = 0; it < 2; ++it) {
          int idx = (it * 128 + tid) * 8;
          int r = idx / GK, c = idx % GK;
          uint4 v = make_uint4(0u, 0u, 0u, 0u);
          if (m0 + r < M) v = *(const uint4*)(Ab + (long)(m0 + r) * K + k0 + c);
          *(uint4*)(&As[r * LDP + c]) = v;
        }
      }
    } else {
      // branch-free im2col gather from halo-padded (Nseq, C, rowL) feature map.
      // k0 % 8 == 0  =>  ci*rowL + tap = (k0>>3)*rowL + (c>>3)*rowL + (c&7),
      // so per thread only a uniform (k0>>3)*rowL term varies across K-steps.
      const int rowL = lout + 7;                  // (Lin = lout-1) + 8 halo
      const int c = tid & 31;                     // column: same for all its
      const int rbase = tid >> 5;
      const int cOff = (c >> 3) * rowL + (c & 7);
      const long kOff = (long)(k0 >> 3) * rowL;
      #pragma unroll 4
      for (int it = 0; it < 16; ++it) {
        int r = rbase + it * 4;
        int m = m0 + r;                           // M multiple of 64: always valid
        int seq = m / lout, l = m - seq * lout;   // const divisor: mul-shift
        As[r * LDP + c] = Ab[(long)seq * cCin * rowL + kOff + cOff + l];
      }
    }

    // ---- stage B tile into LDS transposed: Bs[n][k] ----
    if (BMODE == B_NK) {
#if ASYNC_LDS
      if (fullN) {
        #pragma unroll
        for (int it = 0; it < 2; ++it) {
          int idx = (it * 128 + tid) * 8;
          int r = idx / GK, c = idx % GK;
          __builtin_amdgcn_global_load_async_to_lds_b128(
              (gv4_t)(AS1 void*)(void*)(Bb + (long)(n0 + r) * K + k0 + c),
              (lv4_t)(AS3 void*)(void*)&Bs[r * LDP + c], 0, 0);
        }
        used_async = true;
      } else
#endif
      {
        #pragma unroll
        for (int it = 0; it < 2; ++it) {
          int idx = (it * 128 + tid) * 8;
          int r = idx / GK, c = idx % GK;
          uint4 v = make_uint4(0u, 0u, 0u, 0u);
          if (n0 + r < N) v = *(const uint4*)(Bb + (long)(n0 + r) * K + k0 + c);
          *(uint4*)(&Bs[r * LDP + c]) = v;
        }
      }
    } else {                           // B is KxN: vector row load + LDS transpose
      if (fullN) {
        #pragma unroll
        for (int it = 0; it < 2; ++it) {
          int idx = (it * 128 + tid) * 8;
          int k = idx / GN, c = idx % GN;      // row k0+k, cols n0+c .. +7
          union { uint4 q; half_t h[8]; } v;
          v.q = *(const uint4*)(Bb + (long)(k0 + k) * N + n0 + c);
          #pragma unroll
          for (int u = 0; u < 8; ++u) Bs[(c + u) * LDP + k] = v.h[u];
        }
      } else {                                 // tail tile (only f3: N=96)
        for (int it = 0; it < 16; ++it) {
          int idx = it * 128 + tid;
          int r = idx >> 5, c = idx & 31;      // r = n, c = k
          half_t v = (half_t)0.f;
          if (n0 + r < N) v = Bb[(long)(k0 + c) * N + n0 + r];
          Bs[r * LDP + c] = v;
        }
      }
    }

    if (used_async) wait_async();
    __syncthreads();

    // hint-prefetch next K tile of A (emits global_prefetch_b8)
    if (AMODE == A_DENSE && kk + 1 < nK) {
      int r = tid >> 1;
      if (m0 + r < M)
        __builtin_prefetch(Ab + (long)(m0 + r) * K + k0 + GK + (tid & 1) * 16, 0, 0);
    }

    // ---- fragments per ISA layouts + 2x2 WMMA ----
    #pragma unroll
    for (int i = 0; i < 2; ++i) {
      // A 16x32 f16: lane(m=fm,half=fh): elems 0..7 -> k=8*fh.., 8..15 -> 16+8*fh..
      union { v16h v; uint4 q[2]; } af;
      const half_t* ap = &As[(wm + i * 16 + fm) * LDP + 8 * fh];
      af.q[0] = *(const uint4*)ap;
      af.q[1] = *(const uint4*)(ap + 16);
      #pragma unroll
      for (int j = 0; j < 2; ++j) {
        // B 32x16 f16: lane(n=fm,half=fh): elems 0..7 -> k=8*fh.., 8..15 -> 16+8*fh..
        union { v16h v; uint4 q[2]; } bf;
        const half_t* bp = &Bs[(wn + j * 16 + fm) * LDP + 8 * fh];
        bf.q[0] = *(const uint4*)bp;
        bf.q[1] = *(const uint4*)(bp + 16);
        acc[i][j] = __builtin_amdgcn_wmma_f32_16x16x32_f16(
            false, af.v, false, bf.v, (short)0, acc[i][j], false, false);
      }
    }
    __syncthreads();
  }

  // ---- fused epilogue + store (C/D layout: VGPR r -> row r + 8*fh, col fm) ----
  #pragma unroll
  for (int i = 0; i < 2; ++i) {
    #pragma unroll
    for (int j = 0; j < 2; ++j) {
      const int col = n0 + wn + j * 16 + fm;
      if (col >= N) continue;
      float bv = 0.f;
      if (bias && (ep == EP_BIAS || ep == EP_BIAS_RELU || ep == EP_BIAS_LEAKY)) bv = bias[col];
      #pragma unroll
      for (int r = 0; r < 8; ++r) {
        const int row = m0 + wm + i * 16 + r + 8 * fh;
        if (row >= M) continue;
        float v = acc[i][j][r] + bv;
        if (ep == EP_BIAS_RELU)       v = v > 0.f ? v : 0.f;
        else if (ep == EP_BIAS_LEAKY) v = v > 0.f ? v : 0.01f * v;
        else if (ep == EP_ADJ) {      // leaky(x - 1e8*I) for graph construction
          if (row == col) v -= 1e8f;
          v = v > 0.f ? v : 0.01f * v;
        }
        long off;
        if (conv_store)               // store conv GEMM as (Nseq, Cout, Lout)
          off = ((long)(row / lout) * N + col) * lout + (row % lout);
        else
          off = (long)z * sC + (long)row * N + col;
        if (store_half) ((half_t*)Cv)[off] = (half_t)v;
        else            ((float*)Cv)[off]  = v;
      }
    }
  }
}

template<int AMODE, int BMODE, int CLOUT = 0>
static void gemm(hipStream_t s, const half_t* A, const half_t* B, const float* bias,
                 void* C, int M, int N, int K, long sA, long sB, long sC,
                 int ep, int storeHalf, int batch = 1, int convStore = 0,
                 int cLout = 1, int cCin = 0)
{
  dim3 grid((M + GM - 1) / GM, (N + GN - 1) / GN, batch);
  k_gemm_wmma<AMODE, BMODE, CLOUT><<<grid, dim3(128), 0, s>>>(
      A, B, bias, C, M, N, K, sA, sB, sC, ep, storeHalf, convStore, cLout, cCin);
}

// ---------------------------------------------------------------------------
// Helper kernels (VALU; small fractions of total work)
// ---------------------------------------------------------------------------
__global__ void k_cvt(const float* __restrict__ in, half_t* __restrict__ out, long n) {
  long i = (long)blockIdx.x * blockDim.x + threadIdx.x;
  if (i < n) out[i] = (half_t)in[i];
}

// conv1: Cin=1, k=8, pad=4 -> (Nseq, 64, 65) f16
__global__ void k_conv1(const float* __restrict__ x, const float* __restrict__ w,
                        half_t* __restrict__ out, long Nseq) {
  long i = (long)blockIdx.x * blockDim.x + threadIdx.x;
  long total = Nseq * 64 * 65;
  if (i >= total) return;
  int l = (int)(i % 65); int c = (int)((i / 65) % 64); long s = i / (65 * 64);
  float acc = 0.f;
  #pragma unroll
  for (int t = 0; t < 8; ++t) {
    int li = l + t - 4;
    if (li >= 0 && li < 64) acc += x[s * 64 + li] * w[c * 8 + t];
  }
  out[i] = (half_t)acc;
}

// per-channel sum / sum-of-squares; one block per channel, division-free loops.
template<typename T>
__global__ __launch_bounds__(256) void k_stats(const T* __restrict__ in, float* __restrict__ sums,
    int C, long nOuter, long nInner, long sOuter, long sInner, long sC)
{
  const int c = blockIdx.x;
  __shared__ float ss[256], sq[256];
  float s = 0.f, q = 0.f;
  const T* base = in + (long)c * sC;
  if (nInner == 1) {                       // (R, C) feature layouts
    for (long o = threadIdx.x; o < nOuter; o += 256) {
      float v = (float)base[o * sOuter];
      s += v; q += v * v;
    }
  } else if (nOuter >= 256) {              // conv (N,C,L): threads over N, walk L
    for (long o = threadIdx.x; o < nOuter; o += 256) {
      const T* p = base + o * sOuter;
      for (long ii = 0; ii < nInner; ++ii) {
        float v = (float)p[ii * sInner];
        s += v; q += v * v;
      }
    }
  } else {                                 // hn view: few outers, threads over inner
    for (long o = 0; o < nOuter; ++o) {
      const T* p = base + o * sOuter;
      for (long ii = threadIdx.x; ii < nInner; ii += 256) {
        float v = (float)p[ii * sInner];
        s += v; q += v * v;
      }
    }
  }
  ss[threadIdx.x] = s; sq[threadIdx.x] = q; __syncthreads();
  for (int s2 = 128; s2 > 0; s2 >>= 1) {
    if ((int)threadIdx.x < s2) {
      ss[threadIdx.x] += ss[threadIdx.x + s2];
      sq[threadIdx.x] += sq[threadIdx.x + s2];
    }
    __syncthreads();
  }
  if (threadIdx.x == 0) { sums[c] = ss[0]; sums[C + c] = sq[0]; }
}

// BatchNorm + ReLU + MaxPool1d(k=2,s=2,p=1): (N,C,L) -> halo-padded (N,C,Lp+8),
// 4 zero halo elements each side so the conv im2col gather is branch-free.
__global__ void k_bn_relu_pool(const half_t* __restrict__ in, const float* __restrict__ st,
    const float* __restrict__ g, const float* __restrict__ b, half_t* __restrict__ out,
    long Nseq, int C, int L, int Lp)
{
  long i = (long)blockIdx.x * blockDim.x + threadIdx.x;
  long total = Nseq * C * Lp;
  if (i >= total) return;
  int j = (int)(i % Lp); int c = (int)((i / Lp) % C); long s = i / ((long)Lp * C);
  float E = (float)Nseq * (float)L;
  float mean = st[c] / E, var = st[C + c] / E - (st[c] / E) * (st[c] / E);
  float rs = rsqrtf(var + 1e-5f);
  const half_t* row = in + (s * C + c) * (long)L;
  float best = 0.f;   // post-ReLU values are >= 0; -inf pad can never win
  #pragma unroll
  for (int t = 0; t < 2; ++t) {
    int li = 2 * j - 1 + t;
    if (li >= 0 && li < L) {
      float v = ((float)row[li] - mean) * rs * g[c] + b[c];
      v = v > 0.f ? v : 0.f;
      best = v > best ? v : best;
    }
  }
  half_t* orow = out + (s * C + c) * (long)(Lp + 8);
  orow[4 + j] = (half_t)best;
  if (j == 0)      { orow[0] = orow[1] = orow[2] = orow[3] = (half_t)0.f; }
  if (j == Lp - 1) { orow[Lp+4] = orow[Lp+5] = orow[Lp+6] = orow[Lp+7] = (half_t)0.f; }
}

// channel mean over L=10 (halo-padded rows of 18): (N,32,18)->(N,32)
__global__ void k_chan_mean(const half_t* __restrict__ in, half_t* __restrict__ out, long n) {
  long i = (long)blockIdx.x * blockDim.x + threadIdx.x;
  if (i >= n) return;
  float s = 0.f;
  #pragma unroll
  for (int l = 0; l < 10; ++l) s += (float)in[i * 18 + 4 + l];
  out[i] = (half_t)(s * 0.1f);
}

// feature BatchNorm apply (f32 in), optional leaky, f32 and/or f16 outs
__global__ void k_bn_apply_f32(const float* in, const float* __restrict__ st,
    const float* __restrict__ g, const float* __restrict__ b,
    float* outf, half_t* outh, long R, int Dc, float cnt, int leaky)
{
  long i = (long)blockIdx.x * blockDim.x + threadIdx.x;
  if (i >= R * (long)Dc) return;
  int d = (int)(i % Dc);
  float mean = st[d] / cnt, var = st[Dc + d] / cnt - (st[d] / cnt) * (st[d] / cnt);
  float v = (in[i] - mean) * rsqrtf(var + 1e-5f) * g[d] + b[d];
  if (leaky) v = v > 0.f ? v : 0.01f * v;
  if (outf) outf[i] = v;
  if (outh) outh[i] = (half_t)v;
}

// BN apply over windowed view of a (f16 in, strided batch) -> compact f16 hn
__global__ void k_bn_apply_h(const half_t* __restrict__ in, const float* __restrict__ st,
    const float* __restrict__ g, const float* __restrict__ b, half_t* __restrict__ out,
    int Bn, int Rw, int Dc, long sB, float cnt)
{
  long i = (long)blockIdx.x * blockDim.x + threadIdx.x;
  long total = (long)Bn * Rw * Dc;
  if (i >= total) return;
  int d = (int)(i % Dc); long r = (i / Dc) % Rw; long bb = i / ((long)Dc * Rw);
  float mean = st[d] / cnt, var = st[Dc + d] / cnt - (st[d] / cnt) * (st[d] / cnt);
  float v = ((float)in[bb * sB + r * Dc + d] - mean) * rsqrtf(var + 1e-5f) * g[d] + b[d];
  out[i] = (half_t)v;
}

// row softmax + eye + decay^|ti-tj| mask; one block per (b,n) row
__global__ __launch_bounds__(128) void k_softmax_decay(const float* __restrict__ adj,
                                                       half_t* __restrict__ out, int R)
{
  const long row = blockIdx.x;            // b*R + n
  const int n = (int)(row % R);
  const float* p = adj + row * (long)R;
  __shared__ float red[128];
  const int t = threadIdx.x;
  float mx = -1e30f;
  for (int m = t; m < R; m += 128) mx = fmaxf(mx, p[m]);
  red[t] = mx; __syncthreads();
  for (int s2 = 64; s2 > 0; s2 >>= 1) {
    if (t < s2) red[t] = fmaxf(red[t], red[t + s2]);
    __syncthreads();
  }
  mx = red[0]; __syncthreads();
  float sum = 0.f;
  for (int m = t; m < R; m += 128) sum += __expf(p[m] - mx);
  red[t] = sum; __syncthreads();
  for (int s2 = 64; s2 > 0; s2 >>= 1) {
    if (t < s2) red[t] += red[t + s2];
    __syncthreads();
  }
  const float inv = 1.f / red[0];
  const int ti = n >> 7;                  // ns = 128
  for (int m = t; m < R; m += 128) {
    float v = __expf(p[m] - mx) * inv + (m == n ? 1.f : 0.f);
    int tj = m >> 7;
    int da = ti > tj ? ti - tj : tj - ti;
    float dm = 1.f;
    for (int q = 0; q < da; ++q) dm *= 0.7f;
    out[row * (long)R + m] = (half_t)(v * dm);
  }
}

// mean over time-window rows -> write f16 into concat buffer (stride 512, col off)
__global__ void k_window_pool(const float* __restrict__ in, half_t* __restrict__ out,
                              int Bn, int win, int colOff)
{
  long i = (long)blockIdx.x * blockDim.x + threadIdx.x;
  long total = (long)Bn * 128 * 256;
  if (i >= total) return;
  int d = (int)(i % 256); int n = (int)((i / 256) % 128); long b = i / (256 * 128);
  float s = 0.f;
  for (int w = 0; w < win; ++w)
    s += in[((b * win + w) * 128 + n) * 256 + d];
  out[(b * 128 + n) * 512 + colOff + d] = (half_t)(s / (float)win);
}

// (B,128,96) -> (B,96,128)
__global__ void k_transpose_out(const float* __restrict__ in, float* __restrict__ out) {
  long i = (long)blockIdx.x * blockDim.x + threadIdx.x;
  if (i >= 32L * 96 * 128) return;
  int n = (int)(i % 128); int s = (int)((i / 128) % 96); long b = i / (96 * 128);
  out[i] = in[(b * 128 + n) * 96 + s];
}

// ---------------------------------------------------------------------------
extern "C" void kernel_launch(void* const* d_in, const int* in_sizes, int n_in,
                              void* d_out, int out_size, void* d_ws, size_t ws_size,
                              hipStream_t stream)
{
  (void)in_sizes; (void)out_size; (void)ws_size;
  if (n_in < 36) return;

  const float* x    = (const float*)d_in[0];
  const float* w1   = (const float*)d_in[1];
  const float* bn1g = (const float*)d_in[2];
  const float* bn1b = (const float*)d_in[3];
  const float* w2   = (const float*)d_in[4];
  const float* bn2g = (const float*)d_in[5];
  const float* bn2b = (const float*)d_in[6];
  const float* w3   = (const float*)d_in[7];
  const float* bn3g = (const float*)d_in[8];
  const float* bn3b = (const float*)d_in[9];
  const float* m2w  = (const float*)d_in[10];
  const float* m2b  = (const float*)d_in[11];
  const float* m2bng = (const float*)d_in[12];
  const float* m2bnb = (const float*)d_in[13];
  const float* mp_gcw[2]  = {(const float*)d_in[14], (const float*)d_in[22]};
  const float* mp_gcb[2]  = {(const float*)d_in[15], (const float*)d_in[23]};
  const float* mp_bng[2]  = {(const float*)d_in[16], (const float*)d_in[24]};
  const float* mp_bnb[2]  = {(const float*)d_in[17], (const float*)d_in[25]};
  const float* mp_thw[2]  = {(const float*)d_in[18], (const float*)d_in[26]};
  const float* mp_thb[2]  = {(const float*)d_in[19], (const float*)d_in[27]};
  const float* mp_mbng[2] = {(const float*)d_in[20], (const float*)d_in[28]};
  const float* mp_mbnb[2] = {(const float*)d_in[21], (const float*)d_in[29]};
  const float* f1w = (const float*)d_in[30];
  const float* f1b = (const float*)d_in[31];
  const float* f2w = (const float*)d_in[32];
  const float* f2b = (const float*)d_in[33];
  const float* f3w = (const float*)d_in[34];
  const float* f3b = (const float*)d_in[35];
  float* out = (float*)d_out;

  // ---- workspace carve ----
  uint8_t* W = (uint8_t*)d_ws;
  size_t off = 0;
  auto carve = [&](size_t bytes) -> uint8_t* {
    uint8_t* p = W + off; off += (bytes + 255) & ~(size_t)255; return p;
  };
  half_t* wf2  = (half_t*)carve(65536 * 2);
  half_t* wf3  = (half_t*)carve(32768 * 2);
  half_t* m2wf = (half_t*)carve(16384 * 2);
  half_t* gcf[2] = {(half_t*)carve(262144 * 2), (half_t*)carve(262144 * 2)};
  half_t* thf[2] = {(half_t*)carve(131072 * 2), (half_t*)carve(131072 * 2)};
  half_t* f1f = (half_t*)carve(262144 * 2);
  half_t* f2f = (half_t*)carve(32768 * 2);
  half_t* f3f = (half_t*)carve(6144 * 2);
  float*  st  = (float*)carve(4096);
  uint8_t* A0 = carve(16384L * 128 * 34 * 2);   // 142.6 MB arena (conv outs / MPNN scratch)
  uint8_t* B0 = carve(16384L * 64 * 41 * 2);    // pooled1 (halo-padded) -> pooled3
  uint8_t* B1 = carve(16384L * 128 * 26 * 2);   // pooled2 (halo-padded) -> concat/f-outs
  float*  C0  = (float*)carve(16384L * 512 * 4);
  half_t* aF  = (half_t*)carve(16384L * 512 * 2);
  half_t* meanF = (half_t*)carve(16384L * 32 * 2);

  auto g1 = [](long n) { return dim3((unsigned)((n + 255) / 256)); };

  // ---- cast weights to f16 once ----
  k_cvt<<<g1(65536), 256, 0, stream>>>(w2, wf2, 65536);
  k_cvt<<<g1(32768), 256, 0, stream>>>(w3, wf3, 32768);
  k_cvt<<<g1(16384), 256, 0, stream>>>(m2w, m2wf, 16384);
  for (int p = 0; p < 2; ++p) {
    k_cvt<<<g1(262144), 256, 0, stream>>>(mp_gcw[p], gcf[p], 262144);
    k_cvt<<<g1(131072), 256, 0, stream>>>(mp_thw[p], thf[p], 131072);
  }
  k_cvt<<<g1(262144), 256, 0, stream>>>(f1w, f1f, 262144);
  k_cvt<<<g1(32768), 256, 0, stream>>>(f2w, f2f, 32768);
  k_cvt<<<g1(6144), 256, 0, stream>>>(f3w, f3f, 6144);

  // ================= Feature CNN =================
  // conv1 (K=8, VALU) -> (16384,64,65) f16
  half_t* c1o = (half_t*)A0;
  k_conv1<<<g1(16384L * 64 * 65), 256, 0, stream>>>(x, w1, c1o, 16384);
  k_stats<half_t><<<dim3(64), 256, 0, stream>>>(c1o, st, 64, 16384L, 65L, 64L * 65, 1L, 65L);
  half_t* p1 = (half_t*)B0;   // pooled1 (16384,64,33+8halo)
  k_bn_relu_pool<<<g1(16384L * 64 * 33), 256, 0, stream>>>(c1o, st, bn1g, bn1b, p1, 16384, 64, 65, 33);

  // conv2 as implicit-im2col WMMA GEMM: M=16384*34, N=128, K=512 -> (16384,128,34)
  half_t* c2o = (half_t*)A0;
  gemm<A_CONV, B_NK, 34>(stream, p1, wf2, nullptr, c2o, 16384 * 34, 128, 512,
                         0, 0, 0, EP_NONE, 1, 1, 1, 34, 64);
  k_stats<half_t><<<dim3(128), 256, 0, stream>>>(c2o, st, 128, 16384L, 34L, 128L * 34, 1L, 34L);
  half_t* p2 = (half_t*)B1;   // pooled2 (16384,128,18+8halo)
  k_bn_relu_pool<<<g1(16384L * 128 * 18), 256, 0, stream>>>(c2o, st, bn2g, bn2b, p2, 16384, 128, 34, 18);

  // conv3: M=16384*19, N=32, K=1024 -> (16384,32,19)
  half_t* c3o = (half_t*)A0;
  gemm<A_CONV, B_NK, 19>(stream, p2, wf3, nullptr, c3o, 16384 * 19, 32, 1024,
                         0, 0, 0, EP_NONE, 1, 1, 1, 19, 128);
  k_stats<half_t><<<dim3(32), 256, 0, stream>>>(c3o, st, 32, 16384L, 19L, 32L * 19, 1L, 19L);
  half_t* p3 = (half_t*)B0;   // pooled3 (16384,32,10+8halo)
  k_bn_relu_pool<<<g1(16384L * 32 * 10), 256, 0, stream>>>(c3o, st, bn3g, bn3b, p3, 16384, 32, 19, 10);
  k_chan_mean<<<g1(16384L * 32), 256, 0, stream>>>(p3, meanF, 16384L * 32);

  // m2: (16384,32)x(32,512)+b, then batch-BN -> a f16 (16384,512)
  gemm<A_DENSE, B_KN>(stream, meanF, m2wf, m2b, C0, 16384, 512, 32, 0, 0, 0, EP_BIAS, 0, 1);
  k_stats<float><<<dim3(512), 256, 0, stream>>>(C0, st, 512, 16384L, 1L, 512L, 1L, 1L);
  k_bn_apply_f32<<<g1(16384L * 512), 256, 0, stream>>>(C0, st, m2bng, m2bnb,
      (float*)nullptr, aF, 16384L, 512, 16384.f, 0);

  // ================= MPNN blocks =================
  half_t* mF   = (half_t*)(A0 + 0);
  float*  adjF = (float*) (A0 + (20L << 20));
  half_t* adjH = (half_t*)(A0 + (56L << 20));
  half_t* hnF  = (half_t*)(A0 + (76L << 20));
  half_t* aggF = (half_t*)(A0 + (96L << 20));
  float*  thO  = (float*) (A0 + (116L << 20));
  half_t* concatF = (half_t*)(B1 + 0);

  for (int p = 0; p < 2; ++p) {
    const int Rw = (p == 0) ? 384 : 512;       // window*ns rows (nw==1 for both)
    const int win = (p == 0) ? 3 : 4;
    const long sAa = 4L * 128 * 512;           // a batch stride (sn*ns*D)

    // m = h @ gc_w + gc_b                       (batched 32x[Rw,512,512])
    gemm<A_DENSE, B_KN>(stream, aF, gcf[p], mp_gcb[p], mF, Rw, 512, 512,
                        sAa, 0, (long)Rw * 512, EP_BIAS, 1, 32);
    // adj_raw = leaky(m @ m^T - 1e8*I)          (batched NT GEMM, fused epilogue)
    gemm<A_DENSE, B_NK>(stream, mF, mF, nullptr, adjF, Rw, Rw, 512,
                        (long)Rw * 512, (long)Rw * 512, (long)Rw * Rw, EP_ADJ, 0, 32);
    // softmax + eye + decay mask -> f16 adjacency
    k_softmax_decay<<<dim3(32 * Rw), 128, 0, stream>>>(adjF, adjH, Rw);
    // hn = BN(h) over the windowed view of a
    k_stats<half_t><<<dim3(512), 256, 0, stream>>>(aF, st, 512, 32L, (long)Rw, sAa, 512L, 1L);
    k_bn_apply_h<<<g1((long)32 * Rw * 512), 256, 0, stream>>>(aF, st, mp_bng[p], mp_bnb[p],
        hnF, 32, Rw, 512, sAa, (float)(32 * Rw));
    // agg = adj @ hn                            (batched 32x[Rw,512,Rw])
    gemm<A_DENSE, B_KN>(stream, adjH, hnF, nullptr, aggF, Rw, 512, Rw,
                        (long)Rw * Rw, (long)Rw * 512, (long)Rw * 512, EP_NONE, 1, 32);
    // out = agg @ th_w + th_b
    gemm<A_DENSE, B_KN>(stream, aggF, thf[p], mp_thb[p], thO, Rw, 256, 512,
                        (long)Rw * 512, 0, (long)Rw * 256, EP_BIAS, 0, 32);
    // mbn + leaky (in place), then mean-pool over time window into concat cols
    k_stats<float><<<dim3(256), 256, 0, stream>>>(thO, st, 256, (long)32 * Rw, 1L, 256L, 1L, 1L);
    k_bn_apply_f32<<<g1((long)32 * Rw * 256), 256, 0, stream>>>(thO, st, mp_mbng[p], mp_mbnb[p],
        thO, (half_t*)nullptr, (long)32 * Rw, 256, (float)(32 * Rw), 1);
    k_window_pool<<<g1(32L * 128 * 256), 256, 0, stream>>>(thO, concatF, 32, win, p * 256);
  }

  // ================= Output MLP =================
  half_t* f1o = (half_t*)(B1 + (8L << 20));
  half_t* f2o = (half_t*)(B1 + (16L << 20));
  float*  f3o = (float*) (B1 + (20L << 20));
  gemm<A_DENSE, B_KN>(stream, concatF, f1f, f1b, f1o, 4096, 512, 512, 0, 0, 0, EP_BIAS_RELU, 1, 1);
  gemm<A_DENSE, B_KN>(stream, f1o, f2f, f2b, f2o, 4096, 64, 512, 0, 0, 0, EP_BIAS_RELU, 1, 1);
  gemm<A_DENSE, B_KN>(stream, f2o, f3f, f3b, f3o, 4096, 96, 64, 0, 0, 0, EP_BIAS, 0, 1);
  k_transpose_out<<<g1(32L * 96 * 128), 256, 0, stream>>>(f3o, out);
}
